// Transcriber_51857435132442
// MI455X (gfx1250) — matrix operand order repaired and checked by reading, not compiled, not run on hardware
//
#include <hip/hip_runtime.h>
#include <hip/hip_bf16.h>
#include <math.h>

typedef __attribute__((ext_vector_type(16))) _Float16 v16h;
typedef __attribute__((ext_vector_type(8)))  float    v8f;

#define EPSV 1e-5f
#define TOK 1024     // BATCH * SEQ
#define SEQL 512
#define DM 512
#define DI 1024

// ---------------- workspace layout (float offsets) ----------------
static const size_t OFF_BIGA  = 0;                 // 7,503,872 floats
static const size_t OFF_BIGB  = 7503872;           // 7,503,872 floats
static const size_t OFF_X     = 15007744;          // 524,288
static const size_t OFF_XN    = 15532032;          // 524,288
static const size_t OFF_ACCF  = 16056320;          // 524,288
static const size_t OFF_ACCB  = 16580608;          // 524,288
static const size_t OFF_WT2   = 17104896;          // 9,216  (tap-major conv2 weights)
static const size_t OFF_WT3   = 17114112;          // 18,432 (tap-major conv3 weights)
// aliases inside BIGA (mamba scratch, used after conv stack done)
static const size_t OFF_XZ    = OFF_BIGA;            // 1024*2048
static const size_t OFF_XCONV = OFF_BIGA + 2097152;  // 1024*1024
static const size_t OFF_XDBL  = OFF_BIGA + 3145728;  // 1024*64
static const size_t OFF_DTB   = OFF_BIGA + 3211264;  // 1024*1024
// aliases inside BIGB
static const size_t OFF_YBUF  = OFF_BIGB;            // 1024*1024
static const size_t OFF_GATED = OFF_BIGB + 1048576;  // 1024*1024
static const size_t OFF_FFH   = OFF_BIGB + 2097152;  // 1024*2048

// ================= WMMA fragment loaders =================
// A fragment: 16x32 f16, row m0+(lane&15); lanes 0-15 hold K {0..7,16..23},
// lanes 16-31 hold K {8..15,24..31} (CDNA5 ISA 7.12.2 layout).
__device__ __forceinline__ v16h load_a_frag(const float* __restrict__ A, int lda,
                                            int m0, int k0, int lane) {
  int row = m0 + (lane & 15);
  int kg  = (lane >> 4) << 3;          // 0 or 8
  const float* p = A + (size_t)row * lda + k0;
  v16h a;
#pragma unroll
  for (int j = 0; j < 4; ++j) {
    float2 lo = *(const float2*)(p + kg + 2 * j);
    float2 hi = *(const float2*)(p + 16 + kg + 2 * j);
    a[2*j]       = (_Float16)lo.x; a[2*j+1]     = (_Float16)lo.y;
    a[8 + 2*j]   = (_Float16)hi.x; a[8 + 2*j+1] = (_Float16)hi.y;
  }
  return a;
}
// B fragment: 32x16 f16, column n0+(lane&15); lanes 0-15 hold K {0..7,16..23}, lanes 16-31 hold K {8..15,24..31}.
__device__ __forceinline__ v16h load_b_frag(const float* __restrict__ W, int ldw,
                                            int n0, int k0, int N, int lane) {
  int col = n0 + (lane & 15);
  if (col > N - 1) col = N - 1;        // clamp; stores are guarded
  int kh = (lane >> 4) << 3;           // 0 or 8
  const float* p = W + (size_t)col * ldw + k0 + kh;
  v16h b;
#pragma unroll
  for (int j = 0; j < 4; ++j) {
    float2 lo = *(const float2*)(p + 2 * j);
    float2 hi = *(const float2*)(p + 16 + 2 * j);
    b[2*j] = (_Float16)lo.x; b[2*j+1] = (_Float16)lo.y;
    b[8 + 2*j] = (_Float16)hi.x; b[8 + 2*j+1] = (_Float16)hi.y;
  }
  return b;
}

__device__ __forceinline__ float apply_act(float v, int act) {
  if (act == 1) return 0.5f * v * (1.f + erff(v * 0.70710678118f));        // gelu exact
  if (act == 2) return (v > 20.f) ? v : log1pf(expf(v));                   // softplus
  if (act == 3) return 1.f / (1.f + expf(-v));                             // sigmoid
  return v;
}

// ================= WMMA GEMM : C = act(A[MxK] @ W[NxK]^T + bias) =================
// block = 128 threads (4 waves); each wave computes a 32x32 C tile (2x2 WMMA tiles).
__global__ void wmma_gemm_k(const float* __restrict__ A, const float* __restrict__ W,
                            const float* __restrict__ bias, float* __restrict__ C,
                            int M, int N, int K, int lda, int ldc, int act) {
  const int lane = threadIdx.x & 31;
  const int wave = threadIdx.x >> 5;
  const int m_base = blockIdx.y * 64 + (wave >> 1) * 32;
  const int n_base = blockIdx.x * 64 + (wave & 1) * 32;
  v8f c00 = {}, c01 = {}, c10 = {}, c11 = {};
  for (int k0 = 0; k0 < K; k0 += 32) {
    if (k0 + 32 < K)
      __builtin_prefetch(A + (size_t)(m_base + (lane & 15)) * lda + k0 + 32, 0, 1);
    v16h a0 = load_a_frag(A, lda, m_base,      k0, lane);
    v16h a1 = load_a_frag(A, lda, m_base + 16, k0, lane);
    v16h b0 = load_b_frag(W, K, n_base,      k0, N, lane);
    v16h b1 = load_b_frag(W, K, n_base + 16, k0, N, lane);
    c00 = __builtin_amdgcn_wmma_f32_16x16x32_f16(false, a0, false, b0, (short)0, c00, false, false);
    c01 = __builtin_amdgcn_wmma_f32_16x16x32_f16(false, a0, false, b1, (short)0, c01, false, false);
    c10 = __builtin_amdgcn_wmma_f32_16x16x32_f16(false, a1, false, b0, (short)0, c10, false, false);
    c11 = __builtin_amdgcn_wmma_f32_16x16x32_f16(false, a1, false, b1, (short)0, c11, false, false);
  }
  // C/D layout: VGPR r -> row (lane>=16 ? 8+r : r), col lane&15
  const int rbase = (lane >> 4) << 3;
  const int cc = lane & 15;
#pragma unroll
  for (int mi = 0; mi < 2; ++mi) {
#pragma unroll
    for (int ni = 0; ni < 2; ++ni) {
      v8f cv = (mi == 0) ? ((ni == 0) ? c00 : c01) : ((ni == 0) ? c10 : c11);
      int n = n_base + ni * 16 + cc;
      if (n >= N) continue;
      float bv = bias ? bias[n] : 0.f;
      int m0 = m_base + mi * 16 + rbase;
#pragma unroll
      for (int r = 0; r < 8; ++r) {
        float v = apply_act(cv[r] + bv, act);
        C[(size_t)(m0 + r) * ldc + n] = v;
      }
    }
  }
}

static void gemm(hipStream_t s, const float* A, const float* W, const float* bias,
                 float* C, int M, int N, int K, int lda, int ldc, int act) {
  dim3 g((N + 63) / 64, (M + 63) / 64), b(128);
  wmma_gemm_k<<<g, b, 0, s>>>(A, W, bias, C, M, N, K, lda, ldc, act);
}

// ================= conv stack (NHWC so channels = WMMA K axis) =================
// conv1: mel (2,512,229) Cin=1 -> NHWC out (2,512,229,32) with BN+ReLU (VALU, 9 MACs)
__global__ void conv1_nhwc_k(const float* __restrict__ mel, const float* __restrict__ w,
                             const float* __restrict__ b, const float* __restrict__ g,
                             const float* __restrict__ be, float* __restrict__ out) {
  const int F = 229, T = 512, C = 32;
  int idx = blockIdx.x * blockDim.x + threadIdx.x;  // c fastest
  int total = 2 * T * F * C;
  if (idx >= total) return;
  int c = idx & 31; int f = (idx >> 5) % F; int t = (idx / (C * F)) % T;
  int bI = idx / (C * F * T);
  float acc = 0.f;
#pragma unroll
  for (int dy = 0; dy < 3; ++dy) {
    int tt = t + dy - 1; if (tt < 0 || tt >= T) continue;
#pragma unroll
    for (int dx = 0; dx < 3; ++dx) {
      int ff = f + dx - 1; if (ff < 0 || ff >= F) continue;
      acc += w[c * 9 + dy * 3 + dx] * mel[((size_t)bI * T + tt) * F + ff];
    }
  }
  float scale = g[c] * rsqrtf(1.f + EPSV);
  out[idx] = fmaxf((acc + b[c]) * scale + be[c], 0.f);
}

// weights (Cout,Cin,3,3) -> tap-major (9, Cout, Cin)
__global__ void wtrans_k(const float* __restrict__ w, float* __restrict__ wt,
                         int Cout, int Cin) {
  int idx = blockIdx.x * blockDim.x + threadIdx.x;
  int total = Cout * Cin * 9;
  if (idx >= total) return;
  int tap = idx % 9; int ci = (idx / 9) % Cin; int co = idx / (9 * Cin);
  wt[((size_t)tap * Cout + co) * Cin + ci] = w[idx];
}

// A fragment from NHWC activations at spatial tap (dy,dx); zero at borders.
__device__ __forceinline__ v16h load_conv_a(const float* __restrict__ in, int m0,
                                            int lane, int dy, int dx, int T, int F) {
  int p = m0 + (lane & 15);
  int f = p % F; int t = (p / F) % T; int bI = p / (F * T);
  int tt = t + dy, ff = f + dx;
  v16h a = {};
  if (tt >= 0 && tt < T && ff >= 0 && ff < F) {
    const float* q = in + (((size_t)bI * T + tt) * F + ff) * 32;  // 32 contiguous ch
    int kg = (lane >> 4) << 3;
#pragma unroll
    for (int j = 0; j < 4; ++j) {
      float2 lo = *(const float2*)(q + kg + 2 * j);
      float2 hi = *(const float2*)(q + 16 + kg + 2 * j);
      a[2*j]       = (_Float16)lo.x; a[2*j+1]     = (_Float16)lo.y;
      a[8 + 2*j]   = (_Float16)hi.x; a[8 + 2*j+1] = (_Float16)hi.y;
    }
  }
  return a;
}

// 3x3 conv, Cin=32, NHWC in (2,T,F,32) -> NHWC out (2,T,F,Cout), BN+ReLU.
// 9 taps x one 16x16x32 WMMA each; block = 4 waves, each wave 32 rows x 32 cols.
__global__ void conv_wmma_k(const float* __restrict__ in, const float* __restrict__ wt,
                            const float* __restrict__ b, const float* __restrict__ g,
                            const float* __restrict__ be, float* __restrict__ out,
                            int Cout, int T, int F) {
  const int lane = threadIdx.x & 31;
  const int wave = threadIdx.x >> 5;
  const int m_base = blockIdx.y * 64 + (wave >> 1) * 32;  // flat NHW position
  const int n_base = (wave & 1) * 32;                     // output channel
  v8f c00 = {}, c01 = {}, c10 = {}, c11 = {};
#pragma unroll
  for (int tap = 0; tap < 9; ++tap) {
    int dy = tap / 3 - 1, dx = tap % 3 - 1;
    const float* wtap = wt + (size_t)tap * Cout * 32;
    v16h a0 = load_conv_a(in, m_base,      lane, dy, dx, T, F);
    v16h a1 = load_conv_a(in, m_base + 16, lane, dy, dx, T, F);
    v16h b0 = load_b_frag(wtap, 32, n_base,      0, Cout, lane);
    v16h b1 = load_b_frag(wtap, 32, n_base + 16, 0, Cout, lane);
    c00 = __builtin_amdgcn_wmma_f32_16x16x32_f16(false, a0, false, b0, (short)0, c00, false, false);
    c01 = __builtin_amdgcn_wmma_f32_16x16x32_f16(false, a0, false, b1, (short)0, c01, false, false);
    c10 = __builtin_amdgcn_wmma_f32_16x16x32_f16(false, a1, false, b0, (short)0, c10, false, false);
    c11 = __builtin_amdgcn_wmma_f32_16x16x32_f16(false, a1, false, b1, (short)0, c11, false, false);
  }
  const int rbase = (lane >> 4) << 3;
  const int cc = lane & 15;
#pragma unroll
  for (int mi = 0; mi < 2; ++mi) {
#pragma unroll
    for (int ni = 0; ni < 2; ++ni) {
      v8f cv = (mi == 0) ? ((ni == 0) ? c00 : c01) : ((ni == 0) ? c10 : c11);
      int n = n_base + ni * 16 + cc;
      if (n >= Cout) continue;
      float scale = g[n] * rsqrtf(1.f + EPSV);
      float bb = b[n], bec = be[n];
      int m0 = m_base + mi * 16 + rbase;
#pragma unroll
      for (int r = 0; r < 8; ++r) {
        float v = (cv[r] + bb) * scale + bec;
        out[(size_t)(m0 + r) * Cout + n] = fmaxf(v, 0.f);
      }
    }
  }
}

// NHWC maxpool along F (stride 2): coalesced over contiguous C
__global__ void maxpool_nhwc_k(const float* __restrict__ in, float* __restrict__ out,
                               int C, int T, int Fin) {
  int Fout = Fin >> 1;
  int total = 2 * T * Fout * C;
  int idx = blockIdx.x * blockDim.x + threadIdx.x;
  if (idx >= total) return;
  int c = idx % C; int f = (idx / C) % Fout; int t = (idx / (C * Fout)) % T;
  int bI = idx / (C * Fout * T);
  const float* p = in + (((size_t)bI * T + t) * Fin + 2 * f) * C + c;
  out[idx] = fmaxf(p[0], p[C]);
}

// NHWC (2,512,57,64) -> tokens (1024, 64*57): out[(b*512+t)][c*57+f]
__global__ void transpose_feat_k(const float* __restrict__ in, float* __restrict__ out) {
  int idx = blockIdx.x * blockDim.x + threadIdx.x;
  const int total = 2 * 512 * 57 * 64;
  if (idx >= total) return;
  int c = idx % 64; int f = (idx / 64) % 57; int t = (idx / (64 * 57)) % 512;
  int bI = idx / (64 * 57 * 512);
  out[((size_t)(bI * 512 + t)) * 3648 + c * 57 + f] = in[idx];
}

// ================= layernorm (one wave per 512-wide row) =================
__global__ void ln_k(const float* __restrict__ x, const float* __restrict__ g,
                     const float* __restrict__ b, float* __restrict__ out, int rows) {
  int wv = (blockIdx.x * blockDim.x + threadIdx.x) >> 5;
  int lane = threadIdx.x & 31;
  if (wv >= rows) return;
  const float* row = x + (size_t)wv * DM;
  float s = 0.f, s2 = 0.f;
  for (int i = lane; i < DM; i += 32) { float v = row[i]; s += v; s2 += v * v; }
#pragma unroll
  for (int off = 16; off > 0; off >>= 1) {
    s  += __shfl_xor(s,  off, 32);
    s2 += __shfl_xor(s2, off, 32);
  }
  float mu = s * (1.f / DM);
  float var = s2 * (1.f / DM) - mu * mu;
  float r = rsqrtf(var + EPSV);
  for (int i = lane; i < DM; i += 32)
    out[(size_t)wv * DM + i] = (row[i] - mu) * r * g[i] + b[i];
}

// ================= mamba pieces =================
__global__ void dwconv_silu_k(const float* __restrict__ xz, const float* __restrict__ w,
                              const float* __restrict__ cb, float* __restrict__ out,
                              int reverse) {
  int idx = blockIdx.x * blockDim.x + threadIdx.x;  // over 2*512*1024
  if (idx >= 2 * SEQL * DI) return;
  int d = idx & (DI - 1);
  int t = (idx >> 10) & (SEQL - 1);
  int bI = idx >> 19;
  float acc = cb[d];
#pragma unroll
  for (int j = 0; j < 4; ++j) {
    int tt = reverse ? (t + 3 - j) : (t + j - 3);
    if (tt < 0 || tt >= SEQL) continue;
    acc += w[d * 4 + j] * xz[((size_t)(bI * SEQL + tt)) * 2048 + d];
  }
  float s = acc / (1.f + expf(-acc));
  out[((size_t)(bI * SEQL + t)) * DI + d] = s;
}

// selective scan: one LANE per (channel, state); 16-lane shfl reduce for y.
// wave handles 2 channels; critical path per step = 1 exp + 1 FMA + 4 shfl hops.
__global__ void scan_k(const float* __restrict__ A_log, const float* __restrict__ Dp,
                       const float* __restrict__ xdbl, const float* __restrict__ dtb,
                       const float* __restrict__ xconv, float* __restrict__ ybuf,
                       int reverse) {
  int wv = (blockIdx.x * blockDim.x + threadIdx.x) >> 5;
  int lane = threadIdx.x & 31;
  int ch = wv * 2 + (lane >> 4);        // 0 .. 2047
  int d = ch & (DI - 1);
  int bI = ch >> 10;
  int n = lane & 15;                    // state index
  float An = -expf(A_log[d * 16 + n]);
  float Dd = Dp[d];
  float h = 0.f;
  for (int s = 0; s < SEQL; ++s) {
    int t = reverse ? (SEQL - 1 - s) : s;
    size_t tok = (size_t)bI * SEQL + t;
    float dtv = dtb[tok * DI + d];       // softplus applied in GEMM epilogue
    float xv  = xconv[tok * DI + d];
    float Bn  = xdbl[tok * 64 + 32 + n];
    float Cn  = xdbl[tok * 64 + 48 + n];
    h = expf(dtv * An) * h + dtv * Bn * xv;
    float y = h * Cn;
    y += __shfl_xor(y, 1, 32);
    y += __shfl_xor(y, 2, 32);
    y += __shfl_xor(y, 4, 32);
    y += __shfl_xor(y, 8, 32);
    if (n == 0) ybuf[tok * DI + d] = y + Dd * xv;
  }
}

__global__ void gate_k(const float* __restrict__ ybuf, const float* __restrict__ xz,
                       float* __restrict__ gated) {
  int idx = blockIdx.x * blockDim.x + threadIdx.x;  // over 1024*1024
  if (idx >= TOK * DI) return;
  int d = idx & (DI - 1);
  size_t tok = (size_t)idx >> 10;
  float z = xz[tok * 2048 + DI + d];
  gated[idx] = ybuf[idx] * (z / (1.f + expf(-z)));
}

__global__ void add3_k(const float* a, const float* b, const float* c, float* out, int n) {
  int i = blockIdx.x * blockDim.x + threadIdx.x;
  if (i < n) out[i] = a[i] + b[i] + c[i];
}
__global__ void add2_k(const float* a, const float* b, float* out, int n) {
  int i = blockIdx.x * blockDim.x + threadIdx.x;
  if (i < n) out[i] = a[i] + b[i];
}

// ================= host orchestration =================
struct MambaW {
  const float *A_log, *D, *conv_b, *conv_w, *dt_b, *dt_w, *in_w, *out_w, *x_w;
};
static inline MambaW mamba_at(void* const* din, int base) {
  MambaW m;  // sorted leaf order: A_log, D, conv_b, conv_w, dt_proj_b, dt_proj_w,
             //                    in_proj_w, out_proj_w, x_proj_w
  m.A_log  = (const float*)din[base + 0];
  m.D      = (const float*)din[base + 1];
  m.conv_b = (const float*)din[base + 2];
  m.conv_w = (const float*)din[base + 3];
  m.dt_b   = (const float*)din[base + 4];
  m.dt_w   = (const float*)din[base + 5];
  m.in_w   = (const float*)din[base + 6];
  m.out_w  = (const float*)din[base + 7];
  m.x_w    = (const float*)din[base + 8];
  return m;
}

static void run_mamba(const MambaW& mw, const float* xin, float* yout, int reverse,
                      float* ws, hipStream_t s) {
  float* xz    = ws + OFF_XZ;
  float* xconv = ws + OFF_XCONV;
  float* xdbl  = ws + OFF_XDBL;
  float* dtb   = ws + OFF_DTB;
  float* ybuf  = ws + OFF_YBUF;
  float* gated = ws + OFF_GATED;
  gemm(s, xin, mw.in_w, nullptr, xz, TOK, 2048, DM, DM, 2048, 0);           // in_proj
  dwconv_silu_k<<<(TOK * DI + 255) / 256, 256, 0, s>>>(xz, mw.conv_w, mw.conv_b, xconv, reverse);
  gemm(s, xconv, mw.x_w, nullptr, xdbl, TOK, 64, DI, DI, 64, 0);            // x_proj
  gemm(s, xdbl, mw.dt_w, mw.dt_b, dtb, TOK, DI, 32, 64, DI, 2);             // dt_proj+softplus
  scan_k<<<128, 256, 0, s>>>(mw.A_log, mw.D, xdbl, dtb, xconv, ybuf, reverse);
  gate_k<<<(TOK * DI + 255) / 256, 256, 0, s>>>(ybuf, xz, gated);
  gemm(s, gated, mw.out_w, nullptr, yout, TOK, DM, DI, DI, DM, 0);          // out_proj
}

extern "C" void kernel_launch(void* const* d_in, const int* in_sizes, int n_in,
                              void* d_out, int out_size, void* d_ws, size_t ws_size,
                              hipStream_t stream) {
  (void)in_sizes; (void)n_in; (void)out_size; (void)ws_size;
  float* ws   = (float*)d_ws;
  float* bigA = ws + OFF_BIGA;
  float* bigB = ws + OFF_BIGB;
  float* X    = ws + OFF_X;
  float* XN   = ws + OFF_XN;
  float* ACCF = ws + OFF_ACCF;
  float* ACCB = ws + OFF_ACCB;
  float* WT2  = ws + OFF_WT2;
  float* WT3  = ws + OFF_WT3;

  const float* mel = (const float*)d_in[0];
  // ---- conv params (base 105, sorted): b1 b2 b3 be1 be2 be3 fc_b fc_w g1 g2 g3 w1 w2 w3
  const int CB = 105;
  const float* cb1  = (const float*)d_in[CB + 0];
  const float* cb2  = (const float*)d_in[CB + 1];
  const float* cb3  = (const float*)d_in[CB + 2];
  const float* cbe1 = (const float*)d_in[CB + 3];
  const float* cbe2 = (const float*)d_in[CB + 4];
  const float* cbe3 = (const float*)d_in[CB + 5];
  const float* fc_b = (const float*)d_in[CB + 6];
  const float* fc_w = (const float*)d_in[CB + 7];
  const float* cg1  = (const float*)d_in[CB + 8];
  const float* cg2  = (const float*)d_in[CB + 9];
  const float* cg3  = (const float*)d_in[CB + 10];
  const float* cw1  = (const float*)d_in[CB + 11];
  const float* cw2  = (const float*)d_in[CB + 12];
  const float* cw3  = (const float*)d_in[CB + 13];

  // ---- conv stack (NHWC): mel (2,512,229) -> x (1024, 512)
  {
    wtrans_k<<<(32 * 32 * 9 + 255) / 256, 256, 0, stream>>>(cw2, WT2, 32, 32);
    wtrans_k<<<(64 * 32 * 9 + 255) / 256, 256, 0, stream>>>(cw3, WT3, 64, 32);
    int n1 = 2 * 512 * 229 * 32;
    conv1_nhwc_k<<<(n1 + 255) / 256, 256, 0, stream>>>(mel, cw1, cb1, cg1, cbe1, bigA);
    dim3 gconv2(1, (2 * 512 * 229) / 64);           // 3664 position-tiles
    conv_wmma_k<<<gconv2, 128, 0, stream>>>(bigA, WT2, cb2, cg2, cbe2, bigB, 32, 512, 229);
    int p1 = 2 * 512 * 114 * 32;
    maxpool_nhwc_k<<<(p1 + 255) / 256, 256, 0, stream>>>(bigB, bigA, 32, 512, 229);
    dim3 gconv3(1, (2 * 512 * 114) / 64);           // 1824 position-tiles
    conv_wmma_k<<<gconv3, 128, 0, stream>>>(bigA, WT3, cb3, cg3, cbe3, bigB, 64, 512, 114);
    int p2 = 2 * 512 * 57 * 64;
    maxpool_nhwc_k<<<(p2 + 255) / 256, 256, 0, stream>>>(bigB, bigA, 64, 512, 114);
    transpose_feat_k<<<(p2 + 255) / 256, 256, 0, stream>>>(bigA, bigB);
    gemm(stream, bigB, fc_w, fc_b, X, TOK, DM, 3648, 3648, DM, 0);
  }

  // ---- 4 bidirectional mamba blocks
  for (int blk = 0; blk < 4; ++blk) {
    int base = 1 + 26 * blk;
    MambaW bwd = mamba_at(d_in, base + 0);
    const float* ff_b1   = (const float*)d_in[base + 9];
    const float* ff_b2   = (const float*)d_in[base + 10];
    const float* ff_ln_b = (const float*)d_in[base + 11];
    const float* ff_ln_g = (const float*)d_in[base + 12];
    const float* ff_w1   = (const float*)d_in[base + 13];
    const float* ff_w2   = (const float*)d_in[base + 14];
    MambaW fwd = mamba_at(d_in, base + 15);
    const float* ln_b    = (const float*)d_in[base + 24];
    const float* ln_g    = (const float*)d_in[base + 25];

    ln_k<<<TOK / 8, 256, 0, stream>>>(X, ln_g, ln_b, XN, TOK);
    run_mamba(fwd, XN, ACCF, 0, ws, stream);
    run_mamba(bwd, XN, ACCB, 1, ws, stream);
    int nel = TOK * DM;
    add3_k<<<(nel + 255) / 256, 256, 0, stream>>>(ACCF, ACCB, X, ACCF, nel);  // h
    ln_k<<<TOK / 8, 256, 0, stream>>>(ACCF, ff_ln_g, ff_ln_b, XN, TOK);
    float* FFH = ws + OFF_FFH;
    gemm(stream, XN, ff_w1, ff_b1, FFH, TOK, 2048, DM, DM, 2048, 1);          // gelu
    gemm(stream, FFH, ff_w2, ff_b2, ACCB, TOK, DM, 2048, 2048, DM, 0);
    add2_k<<<(nel + 255) / 256, 256, 0, stream>>>(ACCB, ACCF, X, nel);        // x = ffn + h
  }

  // ---- heads: output order onset, offset, frame, velocity; param bases sorted
  const int head_base[4] = {145, 132, 119, 158};  // onset, offset, frame, velocity
  float* outp = (float*)d_out;
  for (int hIdx = 0; hIdx < 4; ++hIdx) {
    int hb = head_base[hIdx];
    const float* h_b   = (const float*)d_in[hb + 0];
    const float* h_lnb = (const float*)d_in[hb + 1];
    const float* h_lng = (const float*)d_in[hb + 2];
    MambaW hm = mamba_at(d_in, hb + 3);
    const float* h_w   = (const float*)d_in[hb + 12];
    ln_k<<<TOK / 8, 256, 0, stream>>>(X, h_lng, h_lnb, XN, TOK);
    run_mamba(hm, XN, ACCF, 0, ws, stream);
    int act = (hIdx == 3) ? 0 : 3;  // velocity: no sigmoid
    gemm(stream, ACCF, h_w, h_b, outp + (size_t)hIdx * TOK * 88, TOK, 88, DM, DM, 88, act);
  }
}
